// LigerLMHeadDPO_90185723281620
// MI455X (gfx1250) — compile-verified, matrix operand order checked by reading, not run on hardware
//
#include <hip/hip_runtime.h>
#include <hip/hip_bf16.h>
#include <cmath>
#include <stdint.h>

// Problem constants (B=8, T=512, H=2048, V=32000)
#define BT        4096
#define Tseq      512
#define Hdim      2048
#define Vocab     32000
#define NCHUNK    25
#define CHUNK_COLS (Vocab / NCHUNK)    // 1280
#define MT        64                   // rows per workgroup
#define NT        128                  // cols per n-iteration
#define NITER     (CHUNK_COLS / NT)    // 10
#define KT        32                   // WMMA bf16 K
#define KC        64                   // K-chunk staged per pipeline step
#define NKC       (Hdim / KC)          // 32
#define NBUF      3                    // async pipeline depth: prefetch 2 ahead
#define IGNORE_IDX (-100)
#define BETA      0.1f

typedef __attribute__((ext_vector_type(16))) __bf16 v16bf;
typedef __attribute__((ext_vector_type(8)))  float  v8f;

union Frag { uint4 u4[2]; v16bf v; };

// ---- CDNA5 async memory->LDS copy (ASYNCcnt-tracked, no VGPR round-trip) ----
__device__ __forceinline__ void async_ld_b128(const void* gaddr, unsigned lds_off) {
  asm volatile("global_load_async_to_lds_b128 %0, %1, off"
               :: "v"(lds_off), "v"(gaddr) : "memory");
}
__device__ __forceinline__ void wait_async_le8() {
  asm volatile("s_wait_asynccnt 0x8" ::: "memory");
}
__device__ __forceinline__ void wait_async_le4() {
  asm volatile("s_wait_asynccnt 0x4" ::: "memory");
}
__device__ __forceinline__ void wait_async_le0() {
  asm volatile("s_wait_asynccnt 0x0" ::: "memory");
}

__device__ __forceinline__ unsigned int f2bf_rne(float f) {
  union { float f; unsigned int u; } x; x.f = f;
  unsigned int r = x.u + 0x7FFFu + ((x.u >> 16) & 1u);
  return r >> 16;
}

// -------- f32 -> bf16 conversion --------
__global__ __launch_bounds__(256) void k_convert_bf16(
    const float* __restrict__ src, unsigned short* __restrict__ dst, int n8) {
  int i = blockIdx.x * blockDim.x + threadIdx.x;
  if (i >= n8) return;
  const float4* s = (const float4*)src;
  float4 a = s[2 * i], b = s[2 * i + 1];
  uint4 o;
  o.x = (f2bf_rne(a.y) << 16) | f2bf_rne(a.x);
  o.y = (f2bf_rne(a.w) << 16) | f2bf_rne(a.z);
  o.z = (f2bf_rne(b.y) << 16) | f2bf_rne(b.x);
  o.w = (f2bf_rne(b.w) << 16) | f2bf_rne(b.z);
  ((uint4*)dst)[i] = o;
}

// Stage one 128-row x 64-K bf16 B tile (16 KB) via 4 async b128 per thread.
__device__ __forceinline__ void stage_B(const unsigned short* __restrict__ Wk,
                                        unsigned dst_off, int tid) {
#pragma unroll
  for (int j = 0; j < 4; ++j) {
    int idx = j * 256 + tid;            // 0..1023 x 16B
    int row = idx >> 3;                 // 0..127
    int kc  = (idx & 7) * 8;            // 0..56
    async_ld_b128(Wk + (size_t)row * Hdim + kc, dst_off + (unsigned)(row * KC + kc) * 2);
  }
}

// -------- fused GEMM + online softmax over one V-chunk -----------------------
// grid = (BT/MT, NCHUNK), block = 256 (8 waves: 2 wm x 4 wn, 32x32 C per wave)
__global__ __launch_bounds__(256) void k_gemm_softmax(
    const unsigned short* __restrict__ xb,   // BT x H (bf16)
    const unsigned short* __restrict__ Wb,   // V  x H (bf16)
    const int* __restrict__ y,
    float* __restrict__ pmax,
    float* __restrict__ psum,
    float* __restrict__ ptgt) {
  extern __shared__ char smem_raw[];
  unsigned short* As = (unsigned short*)smem_raw;                        // 64x2048 = 256 KB
  unsigned short* Bs = (unsigned short*)(smem_raw + (size_t)MT * Hdim * 2); // 3 x 128x64 = 48 KB
  float* stats = (float*)(smem_raw + (size_t)MT * Hdim * 2 + (size_t)NBUF * NT * KC * 2);

  const int tid  = threadIdx.x;
  const int wave = tid >> 5;
  const int lane = tid & 31;
  const int wm   = wave >> 2;      // 0..1 (32-row stripe)
  const int wn   = wave & 3;       // 0..3 (32-col stripe)
  const int half = lane >> 4;
  const int l15  = lane & 15;
  const int koff = half * 8;

  const int m0    = blockIdx.x * MT;
  const int chunk = blockIdx.y;
  const int c0    = chunk * CHUNK_COLS;

  const unsigned as_off = (unsigned)(size_t)(void*)As;
  const unsigned bs_off = (unsigned)(size_t)(void*)Bs;

  // Async-stage the A tile (contiguous 64 x 2048 bf16 = 256 KB) once.
  {
    const char* src = (const char*)(xb + (size_t)m0 * Hdim);
    const int total = MT * Hdim / 8;
    for (int i = tid; i < total; i += 256)
      async_ld_b128(src + (size_t)i * 16, as_off + (unsigned)i * 16);
    wait_async_le0();
  }
  __syncthreads();

  // Row bookkeeping for the 2 mi sub-tiles (C: M = mi*16 + half*8 + r).
  int ytgt[2][8];
#pragma unroll
  for (int mi = 0; mi < 2; ++mi)
#pragma unroll
    for (int r = 0; r < 8; ++r) {
      int row = m0 + wm * 32 + mi * 16 + half * 8 + r;
      int yv = y[row];
      ytgt[mi][r] = (yv < 0) ? 0 : yv;
    }

  float mrun[2][8], srun[2][8];
#pragma unroll
  for (int mi = 0; mi < 2; ++mi)
#pragma unroll
    for (int r = 0; r < 8; ++r) { mrun[mi][r] = -INFINITY; srun[mi][r] = 0.0f; }

  const unsigned short* a0_row = As + (size_t)(wm * 32 + l15) * Hdim;
  const unsigned short* a1_row = a0_row + (size_t)16 * Hdim;

  for (int it = 0; it < NITER; ++it) {
    const int n0 = c0 + it * NT;
    const unsigned short* Wn = Wb + (size_t)n0 * Hdim;

    stage_B(Wn,      bs_off,                    tid);   // chunk 0 -> buf 0
    stage_B(Wn + KC, bs_off + (unsigned)NT * KC * 2, tid); // chunk 1 -> buf 1

    v8f acc00 = {}, acc01 = {}, acc10 = {}, acc11 = {};

    for (int cch = 0; cch < NKC; ++cch) {
      if (cch + 2 < NKC) {
        stage_B(Wn + (cch + 2) * KC,
                bs_off + (unsigned)(((cch + 2) % NBUF) * NT * KC * 2), tid);
        wait_async_le8();     // chunks c+1,c+2 outstanding -> chunk c done
      } else if (cch + 1 < NKC) {
        wait_async_le4();
      } else {
        wait_async_le0();
      }
      __syncthreads();        // all waves' async writes for chunk c visible

      const unsigned short* bufp = Bs + (size_t)(cch % NBUF) * NT * KC;
      const unsigned short* b0_row = bufp + (size_t)(wn * 32 + l15) * KC;
      const unsigned short* b1_row = b0_row + (size_t)16 * KC;
#pragma unroll
      for (int ks = 0; ks < KC / KT; ++ks) {
        const int ko = ks * KT + koff;
        const int kg = cch * KC + ks * KT + koff;   // global K for A
        Frag a0, a1, b0, b1;
        a0.u4[0] = *(const uint4*)(a0_row + kg);
        a0.u4[1] = *(const uint4*)(a0_row + kg + 16);
        a1.u4[0] = *(const uint4*)(a1_row + kg);
        a1.u4[1] = *(const uint4*)(a1_row + kg + 16);
        b0.u4[0] = *(const uint4*)(b0_row + ko);
        b0.u4[1] = *(const uint4*)(b0_row + ko + 16);
        b1.u4[0] = *(const uint4*)(b1_row + ko);
        b1.u4[1] = *(const uint4*)(b1_row + ko + 16);
        acc00 = __builtin_amdgcn_wmma_f32_16x16x32_bf16(false, a0.v, false, b0.v,
                                                        (short)0, acc00, false, false);
        acc01 = __builtin_amdgcn_wmma_f32_16x16x32_bf16(false, a0.v, false, b1.v,
                                                        (short)0, acc01, false, false);
        acc10 = __builtin_amdgcn_wmma_f32_16x16x32_bf16(false, a1.v, false, b0.v,
                                                        (short)0, acc10, false, false);
        acc11 = __builtin_amdgcn_wmma_f32_16x16x32_bf16(false, a1.v, false, b1.v,
                                                        (short)0, acc11, false, false);
      }
      __syncthreads();        // done reading buf (c % NBUF)
    }

    // Online softmax epilogue over the 4 accumulators.
    const int colb = n0 + wn * 32 + l15;
#pragma unroll
    for (int mi = 0; mi < 2; ++mi) {
      const v8f* accn0 = mi ? &acc10 : &acc00;
      const v8f* accn1 = mi ? &acc11 : &acc01;
      const int rowb = m0 + wm * 32 + mi * 16 + half * 8;
#pragma unroll
      for (int r = 0; r < 8; ++r) {
        float v0 = (*accn0)[r], v1 = (*accn1)[r];
        if (colb == ytgt[mi][r])      ptgt[rowb + r] = v0;
        if (colb + 16 == ytgt[mi][r]) ptgt[rowb + r] = v1;
        float mn = fmaxf(mrun[mi][r], fmaxf(v0, v1));
        srun[mi][r] = srun[mi][r] * __expf(mrun[mi][r] - mn)
                    + __expf(v0 - mn) + __expf(v1 - mn);
        mrun[mi][r] = mn;
      }
    }
  }

  // Merge (max,sum) across the 16 lanes of each half, then across wn stripes.
#pragma unroll
  for (int mi = 0; mi < 2; ++mi)
#pragma unroll
    for (int r = 0; r < 8; ++r) {
      float mm = mrun[mi][r], ss = srun[mi][r];
#pragma unroll
      for (int off = 8; off >= 1; off >>= 1) {
        float mo = __shfl_xor(mm, off, 16);
        float so = __shfl_xor(ss, off, 16);
        float mn = fmaxf(mm, mo);
        ss = ss * __expf(mm - mn) + so * __expf(mo - mn);
        mm = mn;
      }
      if (l15 == 0) {
        int rl = wm * 32 + mi * 16 + half * 8 + r;
        stats[(rl * 4 + wn) * 2 + 0] = mm;
        stats[(rl * 4 + wn) * 2 + 1] = ss;
      }
    }
  __syncthreads();

  if (tid < MT) {
    float M = -INFINITY;
#pragma unroll
    for (int j = 0; j < 4; ++j) M = fmaxf(M, stats[(tid * 4 + j) * 2 + 0]);
    float S = 0.0f;
#pragma unroll
    for (int j = 0; j < 4; ++j)
      S += stats[(tid * 4 + j) * 2 + 1] * __expf(stats[(tid * 4 + j) * 2 + 0] - M);
    int row = m0 + tid;
    pmax[row * NCHUNK + chunk] = M;
    psum[row * NCHUNK + chunk] = S;
  }
}

// -------- combine chunk partials -> token logp -> per-block sums -------------
__global__ __launch_bounds__(256) void k_rowreduce(
    const float* __restrict__ pmax, const float* __restrict__ psum,
    const float* __restrict__ ptgt, const int* __restrict__ y,
    float* __restrict__ blocksum) {
  __shared__ float red[256];
  int tid = threadIdx.x;
  int row = blockIdx.x * 256 + tid;
  float M = -INFINITY;
#pragma unroll
  for (int c = 0; c < NCHUNK; ++c) M = fmaxf(M, pmax[row * NCHUNK + c]);
  float S = 0.0f;
#pragma unroll
  for (int c = 0; c < NCHUNK; ++c)
    S += psum[row * NCHUNK + c] * __expf(pmax[row * NCHUNK + c] - M);
  float lse = M + __logf(S);
  float lp = (y[row] != IGNORE_IDX) ? (ptgt[row] - lse) : 0.0f;
  red[tid] = lp;
  __syncthreads();
  for (int s = 128; s > 0; s >>= 1) {
    if (tid < s) red[tid] += red[tid + s];
    __syncthreads();
  }
  if (tid == 0) blocksum[blockIdx.x] = red[0];
}

// -------- final DPO loss ------------------------------------------------------
__global__ void k_loss(const float* __restrict__ bsum, float* __restrict__ out) {
  float lp[8], rlp[8];
  for (int s = 0; s < 8; ++s) {
    lp[s]  = bsum[2 * s] + bsum[2 * s + 1];
    rlp[s] = bsum[16 + 2 * s] + bsum[16 + 2 * s + 1];
  }
  float loss = 0.0f;
  for (int i = 0; i < 4; ++i) {
    float d = BETA * ((lp[i] - lp[4 + i]) - (rlp[i] - rlp[4 + i]));
    loss += fmaxf(-d, 0.0f) + __logf(1.0f + __expf(-fabsf(d)));
  }
  out[0] = loss * 0.25f;
}

extern "C" void kernel_launch(void* const* d_in, const int* in_sizes, int n_in,
                              void* d_out, int out_size, void* d_ws, size_t ws_size,
                              hipStream_t stream) {
  const float* x     = (const float*)d_in[0];
  const float* ref_x = (const float*)d_in[1];
  const int*   y     = (const int*)d_in[2];
  const float* W     = (const float*)d_in[3];
  const float* ref_W = (const float*)d_in[4];
  float* out = (float*)d_out;

  char* ws = (char*)d_ws;
  size_t off = 0;
  unsigned short* xb = (unsigned short*)(ws + off); off += (size_t)BT * Hdim * 2;
  unsigned short* Wb = (unsigned short*)(ws + off); off += (size_t)Vocab * Hdim * 2;
  float* pmax = (float*)(ws + off); off += (size_t)BT * NCHUNK * 4;
  float* psum = (float*)(ws + off); off += (size_t)BT * NCHUNK * 4;
  float* ptgt = (float*)(ws + off); off += (size_t)BT * 4;
  float* bsum = (float*)(ws + off); off += 32 * 4;

  const dim3 blk(256);
  const int n8x = BT * Hdim / 8;
  const int n8w = Vocab * Hdim / 8;
  const size_t lds = (size_t)MT * Hdim * 2 + (size_t)NBUF * NT * KC * 2
                   + (size_t)MT * 4 * 2 * sizeof(float);   // 313,344 B < 320 KB

  for (int model = 0; model < 2; ++model) {
    const float* xs = model ? ref_x : x;
    const float* Ws = model ? ref_W : W;
    k_convert_bf16<<<(n8x + 255) / 256, blk, 0, stream>>>(xs, xb, n8x);
    k_convert_bf16<<<(n8w + 255) / 256, blk, 0, stream>>>(Ws, Wb, n8w);
    k_gemm_softmax<<<dim3(BT / MT, NCHUNK), blk, lds, stream>>>(xb, Wb, y, pmax, psum, ptgt);
    k_rowreduce<<<BT / 256, blk, 0, stream>>>(pmax, psum, ptgt, y, bsum + model * 16);
  }
  k_loss<<<1, 1, 0, stream>>>(bsum, out);
}
